// MS_G3D_50165218018127
// MI455X (gfx1250) — compile-verified, matrix operand order checked by reading, not run on hardware
//
#include <hip/hip_runtime.h>

// ---------------------------------------------------------------------------
// MS-G3D fused forward for gfx1250 (MI455X), bf16 WMMA pipeline.
// ---------------------------------------------------------------------------

typedef __bf16        v16bf __attribute__((ext_vector_type(16)));
typedef float         v8f   __attribute__((ext_vector_type(8)));
typedef unsigned int  v4u   __attribute__((ext_vector_type(4)));
typedef unsigned int  v8u   __attribute__((ext_vector_type(8)));

#define N_    8
#define C_    96
#define T_    128
#define V_    25
#define VL_   75
#define NS_   6
#define HS_   12          // 2*NS slices of 96 input channels each
#define DIN_  1152
#define XUP_  104         // LDS pitch (halves) for xu rows (K padded 75->96)
#define AP_   104         // LDS pitch (halves) for a-slice rows (c padded 96)

// fragment counts (each fragment = 512 bf16 halves, lane-major pre-swizzled)
#define WM_FRAGS  216     // 6 mt * 36 kc   (W_mlp as A-operand)
#define G_FRAGS   180     // 12 hs * 5 vt * 3 kc (graph matrices as B-operand)
#define WO_FRAGS  54      // 6 mt * 9 kc    (W_out as A-operand)
#define FRAG_HALVES ((WM_FRAGS + G_FRAGS + WO_FRAGS) * 512)

// workspace byte offsets
#define OFF_FRAGS 0
#define OFF_MLPR  ((size_t)FRAG_HALVES * 2)                       //   460,800
#define MLPR_HALVES ((size_t)N_ * C_ * T_ * 80)                   // 7,864,320
#define OFF_COEF1 (OFF_MLPR + MLPR_HALVES * 2)                    // 16,189,440
#define OFF_COEF2 (OFF_COEF1 + 192 * 4)

// ---------------------------------------------------------------------------
__device__ __forceinline__ unsigned short f2bf(float f) {
    unsigned u = __float_as_uint(f);
    u += 0x7FFFu + ((u >> 16) & 1u);           // round-to-nearest-even
    return (unsigned short)(u >> 16);
}
__device__ __forceinline__ float bf2f(unsigned short h) {
    return __uint_as_float(((unsigned)h) << 16);
}
// k index inside a 32-wide K-chunk for lane l, half-slot j (A and B layouts)
__device__ __forceinline__ int kmap(int l, int j) {
    return ((l & 16) >> 1) + (j & 7) + ((j & 8) << 1);
}
__device__ __forceinline__ v8f vzero() {
    v8f z;
#pragma unroll
    for (int i = 0; i < 8; ++i) z[i] = 0.0f;
    return z;
}
__device__ __forceinline__ v16bf combine16(v4u lo, v4u hi) {
    v8u q;
    q[0] = lo[0]; q[1] = lo[1]; q[2] = lo[2]; q[3] = lo[3];
    q[4] = hi[0]; q[5] = hi[1]; q[6] = hi[2]; q[7] = hi[3];
    return __builtin_bit_cast(v16bf, q);
}
// pre-swizzled fragment in global memory: 32 lanes * 16 halves contiguous
__device__ __forceinline__ v16bf load_frag_g(const unsigned short* base, int lane) {
    const v4u* p = (const v4u*)(base + lane * 16);
    return combine16(p[0], p[1]);
}
// A-style fragment from row-major LDS tile: row = lane&15, two b128 ds loads
__device__ __forceinline__ v16bf load_frag_lds(const unsigned short* row_base,
                                               int pitch, int k0, int lane) {
    const unsigned short* r = row_base + (lane & 15) * pitch + k0 + ((lane & 16) >> 1);
    v4u lo = *(const v4u*)(r);
    v4u hi = *(const v4u*)(r + 16);
    return combine16(lo, hi);
}
__device__ __forceinline__ v8f wmma_bf16(v16bf a, v16bf b, v8f c) {
    return __builtin_amdgcn_wmma_f32_16x16x32_bf16(false, a, false, b,
                                                   (short)0, c, false, false);
}

// ---------------------------------------------------------------------------
// Kernel 1: build pre-swizzled bf16 fragments for W_mlp, (A|B)+res, W_out.
// ---------------------------------------------------------------------------
__global__ __launch_bounds__(256) void prep_frags(
    const float* __restrict__ As, const float* __restrict__ Bs,
    const float* __restrict__ Ares, const float* __restrict__ Wm,
    const float* __restrict__ Wo, unsigned short* __restrict__ frags)
{
    int gid = blockIdx.x * 256 + threadIdx.x;
    if (gid >= FRAG_HALVES) return;
    int f = gid >> 9, r = gid & 511, l = r >> 4, j = r & 15;
    int k  = kmap(l, j);
    int mn = l & 15;
    float val = 0.0f;
    if (f < WM_FRAGS) {                         // W_mlp A-frag: f = mt*36 + kc
        int mt = f / 36, kc = f % 36;
        val = Wm[(mt * 16 + mn) * DIN_ + kc * 32 + k];
    } else if (f < WM_FRAGS + G_FRAGS) {        // graph B-frag
        int f2 = f - WM_FRAGS;
        int hs = f2 / 15, vt = (f2 % 15) / 3, kc = f2 % 3;
        int v = vt * 16 + mn, u = kc * 32 + k;
        if (v < VL_ && u < VL_) {
            int row = (hs % NS_) * VL_ + v;
            const float* M = (hs < NS_) ? As : Bs;
            val = M[row * VL_ + u] + Ares[row * VL_ + u];
        }
    } else {                                    // W_out A-frag: f = mt*9 + kc
        int f2 = f - WM_FRAGS - G_FRAGS;
        int mt = f2 / 9, kc = f2 % 9;
        int o = mt * 16 + mn, kk = kc * 32 + k;
        int w = kk / 96, i = kk % 96;
        val = Wo[(o * 96 + i) * 3 + w];
    }
    frags[gid] = f2bf(val);
}

// ---------------------------------------------------------------------------
// Kernel 2: fused graph-aggregation + MLP GEMMs. One workgroup per (n,t).
//   stage1: a[c, hs-slice v] = xu(96x96) @ Gslice^T   (WMMA bf16)
//   stage2: mlp[o, v] += Wm[:, hs*96:+96] @ a_slice   (WMMA bf16, reg accum)
// Epilogue: relu(mlp + b_mlp) -> bf16 mlp_r[n][o][t][v] (pitch 80).
// ---------------------------------------------------------------------------
__global__ __launch_bounds__(256) void fused_gemm12(
    const float* __restrict__ x, const float* __restrict__ b_mlp,
    const unsigned short* __restrict__ wmF, const unsigned short* __restrict__ gF,
    unsigned short* __restrict__ mlpr)
{
    __shared__ unsigned short xu[C_ * XUP_];          // 19,968 B
    __shared__ unsigned short ab[2][80 * AP_];        // 33,280 B (ping-pong)

    const int blk = blockIdx.x;
    const int n = blk >> 7, t = blk & 127;
    const int tid = threadIdx.x, lane = tid & 31, wid = tid >> 5;

    // Build temporal-unfold tile xu[c][u], u = w*25+vv, K padded to 96 w/ zeros.
    for (int e = tid; e < C_ * 96; e += 256) {
        int c = e / 96, u = e % 96;
        float v = 0.0f;
        if (u < VL_) {
            int w = u / 25, vv = u % 25, ts = t + w - 1;
            if (ts >= 0 && ts < T_) v = x[((n * C_ + c) * T_ + ts) * V_ + vv];
        }
        xu[c * XUP_ + u] = f2bf(v);
    }
    __syncthreads();

    v8f acc2[4];
#pragma unroll
    for (int s = 0; s < 4; ++s) acc2[s] = vzero();

    for (int hs = 0; hs < HS_; ++hs) {
        unsigned short* buf = ab[hs & 1];
        if (hs + 1 < HS_)   // warm L2/L1 for next slice's graph fragments
            __builtin_prefetch(gF + (size_t)((hs + 1) * 15) * 512 + lane * 16, 0, 1);

        // ---- stage 1: 30 output tiles (6 mt x 5 vt) over 8 waves ----
        for (int idx = wid; idx < 30; idx += 8) {
            int mt = idx / 5, vt = idx % 5;
            v8f a = vzero();
#pragma unroll
            for (int kc = 0; kc < 3; ++kc) {
                v16bf af = load_frag_lds(xu + mt * 16 * XUP_, XUP_, kc * 32, lane);
                v16bf bf = load_frag_g(gF + (size_t)((hs * 5 + vt) * 3 + kc) * 512, lane);
                a = wmma_bf16(af, bf, a);
            }
            // D tile -> LDS transposed as a[v][c] (8 contiguous c per lane)
            int vrow = vt * 16 + (lane & 15);
            int cbase = mt * 16 + ((lane & 16) >> 1);
            unsigned short h[8];
#pragma unroll
            for (int r2 = 0; r2 < 8; ++r2) h[r2] = f2bf(a[r2]);
            v4u pk;
            pk[0] = (unsigned)h[0] | ((unsigned)h[1] << 16);
            pk[1] = (unsigned)h[2] | ((unsigned)h[3] << 16);
            pk[2] = (unsigned)h[4] | ((unsigned)h[5] << 16);
            pk[3] = (unsigned)h[6] | ((unsigned)h[7] << 16);
            *(v4u*)(buf + vrow * AP_ + cbase) = pk;
        }
        __syncthreads();

        // ---- stage 2: accumulate mlp tiles from this slice ----
        int slot = 0;
        for (int idx = wid; idx < 30; idx += 8, ++slot) {
            int mt = idx / 5, vt = idx % 5;
#pragma unroll
            for (int kcc = 0; kcc < 3; ++kcc) {
                v16bf af = load_frag_g(wmF + (size_t)(mt * 36 + hs * 3 + kcc) * 512, lane);
                v16bf bf = load_frag_lds(buf + vt * 16 * AP_, AP_, kcc * 32, lane);
                acc2[slot] = wmma_bf16(af, bf, acc2[slot]);
            }
        }
    }

    // ---- epilogue: bias + ReLU, store bf16 (only v < 75) ----
    int slot = 0;
    for (int idx = wid; idx < 30; idx += 8, ++slot) {
        int mt = idx / 5, vt = idx % 5;
        int v = vt * 16 + (lane & 15);
        if (v < VL_) {
            int obase = mt * 16 + ((lane & 16) >> 1);
#pragma unroll
            for (int r2 = 0; r2 < 8; ++r2) {
                int o = obase + r2;
                float val = fmaxf(acc2[slot][r2] + b_mlp[o], 0.0f);
                mlpr[(size_t)((n * C_ + o) * T_ + t) * 80 + v] = f2bf(val);
            }
        }
    }
}

// ---------------------------------------------------------------------------
// Kernel 3: BN1 batch stats (over relu'd mlp) -> per-channel scale/shift.
// ---------------------------------------------------------------------------
__global__ __launch_bounds__(256) void bn_stats1(
    const unsigned short* __restrict__ mlpr, const float* __restrict__ g1,
    const float* __restrict__ bt1, float* __restrict__ coef)
{
    __shared__ float s1[256], s2[256];
    int ch = blockIdx.x, tid = threadIdx.x;
    float a = 0.f, b = 0.f;
    for (int e = tid; e < N_ * T_ * VL_; e += 256) {
        int n = e / (T_ * VL_), rem = e % (T_ * VL_);
        int t = rem / VL_, v = rem % VL_;
        float r = bf2f(mlpr[(size_t)((n * C_ + ch) * T_ + t) * 80 + v]);
        a += r; b += r * r;
    }
    s1[tid] = a; s2[tid] = b; __syncthreads();
    for (int off = 128; off > 0; off >>= 1) {
        if (tid < off) { s1[tid] += s1[tid + off]; s2[tid] += s2[tid + off]; }
        __syncthreads();
    }
    if (tid == 0) {
        const float inv = 1.0f / (float)(N_ * T_ * VL_);
        float m = s1[0] * inv, var = s2[0] * inv - m * m;
        float sc = g1[ch] * rsqrtf(var + 1e-5f);
        coef[ch] = sc; coef[96 + ch] = bt1[ch] - m * sc;
    }
}

// ---------------------------------------------------------------------------
// Kernel 4: out_conv GEMM (K=288) via WMMA. One workgroup per (n, 8-t slab).
//   y = relu(bn1(relu_mlp)) staged into LDS as bf16, Out = WoK @ Y + b_out.
// ---------------------------------------------------------------------------
__global__ __launch_bounds__(256) void stage3_conv(
    const unsigned short* __restrict__ mlpr, const float* __restrict__ coef1,
    const unsigned short* __restrict__ woF, const float* __restrict__ b_out,
    float* __restrict__ out)
{
    extern __shared__ unsigned short ylds[];   // 96 * 8 * 80 halves = 122,880 B
    const int blk = blockIdx.x;
    const int n = blk >> 4, t0 = (blk & 15) * 8;
    const int tid = threadIdx.x, lane = tid & 31, wid = tid >> 5;

    for (int e = tid; e < C_ * 8 * 80; e += 256) {
        int i = e / 640, rem = e % 640, tt = rem / 80, vp = rem % 80;
        float y = 0.0f;
        if (vp < VL_) {
            float r = bf2f(mlpr[(size_t)((n * C_ + i) * T_ + (t0 + tt)) * 80 + vp]);
            y = fmaxf(r * coef1[i] + coef1[96 + i], 0.0f);
        }
        ylds[e] = f2bf(y);
    }
    __syncthreads();

    for (int pt = wid; pt < 13; pt += 8) {     // 13 tiles over p = t8*25+v (200)
        v8f acc[6];
#pragma unroll
        for (int m = 0; m < 6; ++m) acc[m] = vzero();
        int p = pt * 16 + (lane & 15);
        bool pv = p < 200;
        int tt = pv ? p / 25 : 0;
        int vv = pv ? p % 25 : 0;

        for (int kc = 0; kc < 9; ++kc) {
            int w = kc / 3;
            int ibase = (kc % 3) * 32 + ((lane & 16) >> 1);
            int yoff = tt * 80 + w * 25 + vv;
            v8u q;
#pragma unroll
            for (int jj = 0; jj < 4; ++jj) {
                unsigned a0 = pv ? ylds[(ibase + 2 * jj) * 640 + yoff] : 0u;
                unsigned a1 = pv ? ylds[(ibase + 2 * jj + 1) * 640 + yoff] : 0u;
                unsigned b0 = pv ? ylds[(ibase + 16 + 2 * jj) * 640 + yoff] : 0u;
                unsigned b1 = pv ? ylds[(ibase + 17 + 2 * jj) * 640 + yoff] : 0u;
                q[jj] = a0 | (a1 << 16);
                q[4 + jj] = b0 | (b1 << 16);
            }
            v16bf bfrag = __builtin_bit_cast(v16bf, q);
#pragma unroll
            for (int mt = 0; mt < 6; ++mt) {
                v16bf af = load_frag_g(woF + (size_t)(mt * 9 + kc) * 512, lane);
                acc[mt] = wmma_bf16(af, bfrag, acc[mt]);
            }
        }
        if (pv) {
            int t = t0 + p / 25, v2 = p % 25;
#pragma unroll
            for (int mt = 0; mt < 6; ++mt) {
                int obase = mt * 16 + ((lane & 16) >> 1);
#pragma unroll
                for (int r2 = 0; r2 < 8; ++r2) {
                    int o = obase + r2;
                    out[((n * C_ + o) * T_ + t) * V_ + v2] = acc[mt][r2] + b_out[o];
                }
            }
        }
    }
}

// ---------------------------------------------------------------------------
// Kernel 5: BN2 batch stats over pre-norm output (in d_out).
// ---------------------------------------------------------------------------
__global__ __launch_bounds__(256) void bn_stats2(
    const float* __restrict__ out, const float* __restrict__ g2,
    const float* __restrict__ bt2, float* __restrict__ coef)
{
    __shared__ float s1[256], s2[256];
    int ch = blockIdx.x, tid = threadIdx.x;
    float a = 0.f, b = 0.f;
    for (int e = tid; e < N_ * T_ * V_; e += 256) {
        int n = e / (T_ * V_), rem = e % (T_ * V_);
        int t = rem / V_, v = rem % V_;
        float r = out[((n * C_ + ch) * T_ + t) * V_ + v];
        a += r; b += r * r;
    }
    s1[tid] = a; s2[tid] = b; __syncthreads();
    for (int off = 128; off > 0; off >>= 1) {
        if (tid < off) { s1[tid] += s1[tid + off]; s2[tid] += s2[tid + off]; }
        __syncthreads();
    }
    if (tid == 0) {
        const float inv = 1.0f / (float)(N_ * T_ * V_);
        float m = s1[0] * inv, var = s2[0] * inv - m * m;
        float sc = g2[ch] * rsqrtf(var + 1e-5f);
        coef[ch] = sc; coef[96 + ch] = bt2[ch] - m * sc;
    }
}

// Kernel 6: apply BN2 in place.
__global__ __launch_bounds__(256) void finalize_bn2(
    float* __restrict__ out, const float* __restrict__ coef)
{
    int i = blockIdx.x * 256 + threadIdx.x;
    if (i < N_ * C_ * T_ * V_) {
        int c = (i / (T_ * V_)) % C_;
        out[i] = out[i] * coef[c] + coef[96 + c];
    }
}

// ---------------------------------------------------------------------------
extern "C" void kernel_launch(void* const* d_in, const int* in_sizes, int n_in,
                              void* d_out, int out_size, void* d_ws, size_t ws_size,
                              hipStream_t stream) {
    (void)in_sizes; (void)n_in; (void)out_size; (void)ws_size;
    const float* x      = (const float*)d_in[0];
    const float* As     = (const float*)d_in[1];
    const float* Bs     = (const float*)d_in[2];
    const float* Ares   = (const float*)d_in[3];
    const float* Wm     = (const float*)d_in[4];
    const float* b_mlp  = (const float*)d_in[5];
    const float* g1     = (const float*)d_in[6];
    const float* bt1    = (const float*)d_in[7];
    const float* Wo     = (const float*)d_in[8];
    const float* b_out  = (const float*)d_in[9];
    const float* g2     = (const float*)d_in[10];
    const float* bt2    = (const float*)d_in[11];

    char* ws = (char*)d_ws;
    unsigned short* frags = (unsigned short*)(ws + OFF_FRAGS);
    unsigned short* mlpr  = (unsigned short*)(ws + OFF_MLPR);
    float* coef1 = (float*)(ws + OFF_COEF1);
    float* coef2 = (float*)(ws + OFF_COEF2);
    const unsigned short* wmF = frags;
    const unsigned short* gF  = frags + (size_t)WM_FRAGS * 512;
    const unsigned short* woF = frags + (size_t)(WM_FRAGS + G_FRAGS) * 512;
    float* out = (float*)d_out;

    prep_frags<<<FRAG_HALVES / 256, 256, 0, stream>>>(As, Bs, Ares, Wm, Wo, frags);
    fused_gemm12<<<N_ * T_, 256, 0, stream>>>(x, b_mlp, wmF, gF, mlpr);
    bn_stats1<<<C_, 256, 0, stream>>>(mlpr, g1, bt1, coef1);
    stage3_conv<<<N_ * 16, 256, C_ * 8 * 80 * sizeof(unsigned short), stream>>>(
        mlpr, coef1, woF, b_out, out);
    bn_stats2<<<C_, 256, 0, stream>>>(out, g2, bt2, coef2);
    finalize_bn2<<<(N_ * C_ * T_ * V_ + 255) / 256, 256, 0, stream>>>(out, coef2);
}
